// Group_query_Attention_With_RoPE_20495583936571
// MI455X (gfx1250) — matrix operand order repaired and checked by reading, not compiled, not run on hardware
//
#include <hip/hip_runtime.h>

// ---------------------------------------------------------------------------
// GQA + RoPE for MI455X (gfx1250, wave32, WMMA + TDM).
// All matmuls in bf16 via v_wmma_f32_16x16x32_bf16. GEMM weight tiles are
// DMA'd into LDS with tensor_load_to_lds (TENSORcnt) and double-buffered;
// flash-attention streaming softmax so the T x T score matrix is never
// materialized.
// ---------------------------------------------------------------------------

#define B_   2
#define T_   2048
#define C_   2048
#define KVC_ 512
#define HQ_  32
#define HKV_ 8
#define D_   64
#define G_   4

typedef __bf16 bf16;
typedef __attribute__((ext_vector_type(16))) __bf16 v16bf;
typedef __attribute__((ext_vector_type(8)))  __bf16 v8bf;
typedef __attribute__((ext_vector_type(8)))  float  v8f;
typedef __attribute__((ext_vector_type(4)))  unsigned int v4u;
typedef __attribute__((ext_vector_type(8)))  int v8i;
typedef __attribute__((ext_vector_type(4)))  int v4i;

union V16 { v16bf v; v8bf h[2]; };

__device__ __forceinline__ bf16 f2bf(float f) {
  unsigned u = __float_as_uint(f);
  u += 0x7FFFu + ((u >> 16) & 1u);            // round-to-nearest-even
  unsigned short hs = (unsigned short)(u >> 16);
  union { unsigned short s; bf16 b; } cv; cv.s = hs;
  return cv.b;
}

// Low 32 bits of a generic pointer into the LDS aperture == LDS byte offset.
__device__ __forceinline__ unsigned ldsOffset(const void* p) {
  return (unsigned)(size_t)p;
}

// ------------------- TDM: DMA a [tileN x 32] bf16 tile to LDS --------------
// D# per CDNA5 ISA ch.8: group0 = {count|flags, lds_addr, global_addr, type},
// group1 = {mask/data_size, tensor dims/strides, tile dims}. 2-D tile, so
// the remaining groups are zero. Issued by one wave; completion via TENSORcnt.
// This toolchain (clang-23 / therock-10.0) uses the 6-arg builtin form.
__device__ __forceinline__ void tdm_load_wtile(const bf16* gsrc, unsigned ldsOff,
                                               int K, int N, int tileN) {
  unsigned long long ga = (unsigned long long)(size_t)gsrc;
  v4u g0;
  g0[0] = 1u;                                              // count=1 (valid)
  g0[1] = ldsOff;                                          // lds_addr (bytes)
  g0[2] = (unsigned)ga;                                    // global_addr[31:0]
  g0[3] = ((unsigned)(ga >> 32) & 0x01FFFFFFu) | (2u << 30); // addr[56:32]|type=2
  v8i g1;
  g1[0] = 0x00010000;                                      // data_size=1 -> 2B
  g1[1] = (int)(((unsigned)K & 0xFFFFu) << 16);            // tensor_dim0[15:0]
  g1[2] = (int)(((unsigned)K >> 16) | (((unsigned)N & 0xFFFFu) << 16)); // dim0 hi | dim1 lo
  g1[3] = (int)(((unsigned)N >> 16) | (32u << 16));        // dim1 hi | tile_dim0=32
  g1[4] = tileN;                                           // tile_dim1 (tile_dim2=0)
  g1[5] = K;                                               // tensor_dim0_stride lo
  g1[6] = 0;                                               // stride hi / dim1_stride lo
  g1[7] = 0;
  v4i gz4 = {0, 0, 0, 0};
  v8i gz8 = {0, 0, 0, 0, 0, 0, 0, 0};
  __builtin_amdgcn_tensor_load_to_lds(g0, g1, gz4, gz4, gz8, 0);
}

// --------------------------- f32 -> bf16 convert ---------------------------
__global__ void cvt_f32_bf16(const float* __restrict__ in,
                             bf16* __restrict__ out, int n) {
  int i = blockIdx.x * blockDim.x + threadIdx.x;
  if (i < n) out[i] = f2bf(in[i]);
}

// -------------------- RoPE (interleaved pairs) + convert -------------------
__global__ void rope_cvt(const float* __restrict__ in,
                         bf16* __restrict__ out, int rows, int cols) {
  int pairsPerRow = cols >> 1;
  int idx = blockIdx.x * blockDim.x + threadIdx.x;
  if (idx >= rows * pairsPerRow) return;
  int row = idx / pairsPerRow;
  int p   = idx - row * pairsPerRow;
  int t   = row % T_;                  // position in sequence
  int i   = p & 31;                    // pair index within head (D/2 = 32)
  float inv = __expf(-9.210340371976184f * ((float)i * (1.0f / 32.0f)));
  float ang = (float)t * inv;
  float s, c;
  __sincosf(ang, &s, &c);
  size_t base = (size_t)row * cols + 2 * p;
  float xr = in[base], xi = in[base + 1];
  out[base]     = f2bf(xr * c - xi * s);
  out[base + 1] = f2bf(xr * s + xi * c);
}

// ------------------------- GEMM: C[MxN] = A[MxK] * W[NxK]^T ----------------
// 256 threads = 8 waves; block tile 128x64. The 64x32 W tile (shared by all
// 8 waves) is DMA'd into LDS by the TDM, double-buffered: the DMA for tile
// t+1 is in flight while tile t feeds the WMMAs. A fragments are per-wave
// unique, read straight from global with b128 loads (LOADcnt overlaps
// TENSORcnt).
__global__ __launch_bounds__(256) void gemm_xWt_bf16(
    const bf16* __restrict__ A, const bf16* __restrict__ W,
    float* __restrict__ Cout, int M, int N, int K) {
  __shared__ bf16 Wt[2][64 * 32];               // double-buffered tile, 8 KB

  const int lane = threadIdx.x & 31;
  const int wave = threadIdx.x >> 5;
  const bool lo  = lane < 16;
  const int ln   = lane & 15;
  const int mBase = blockIdx.x * 128 + wave * 16;
  const int nBase = blockIdx.y * 64;
  const bf16* Arow  = A + (size_t)(mBase + ln) * K;
  const bf16* Wtile = W + (size_t)nBase * K;

  if (wave == 0) {
    tdm_load_wtile(Wtile, ldsOffset(&Wt[0][0]), K, N, 64);
    __builtin_amdgcn_s_wait_tensorcnt(0);
  }
  __syncthreads();

  v8f acc[4] = {};
  int cur = 0;
  for (int k0 = 0; k0 < K; k0 += 32, cur ^= 1) {
    if (wave == 0 && k0 + 32 < K)               // kick DMA for next tile
      tdm_load_wtile(Wtile + (k0 + 32), ldsOffset(&Wt[cur ^ 1][0]), K, N, 64);

    if (k0 + 256 < K)
      __builtin_prefetch(Arow + k0 + 256, 0, 3);   // near-cache A prefetch

    V16 a;  // A fragment 16x32: lanes<16 hold K 0-15, lanes>=16 16-31
    a.h[0] = *(const v8bf*)(Arow + k0 + (lo ? 0 : 16));
    a.h[1] = *(const v8bf*)(Arow + k0 + (lo ? 8 : 24));

    const bf16* wt = &Wt[cur][0];
    #pragma unroll
    for (int nt = 0; nt < 4; ++nt) {
      v16bf b = *(const v16bf*)(wt + (nt * 16 + ln) * 32 + (lo ? 0 : 16));
      acc[nt] = __builtin_amdgcn_wmma_f32_16x16x32_bf16(
          false, a.v, false, b, (short)0, acc[nt], false, false);
    }

    __syncthreads();                            // everyone done reading cur
    if (wave == 0) __builtin_amdgcn_s_wait_tensorcnt(0);
    __syncthreads();                            // next buffer visible to all
  }

  #pragma unroll
  for (int nt = 0; nt < 4; ++nt) {
    const int n = nBase + nt * 16 + ln;
    #pragma unroll
    for (int r = 0; r < 8; ++r) {
      int m = mBase + r + (lo ? 0 : 8);          // C/D layout
      Cout[(size_t)m * N + n] = acc[nt][r];
    }
  }
}

// ------------------------------ Flash GQA ----------------------------------
// Block = (b, kv-head, 16-query tile). 4 waves = the 4 q-heads sharing this
// KV head; V tiles (32 keys) are staged once in LDS and shared; K is read from global. Online softmax.
__global__ __launch_bounds__(128) void flash_gqa(
    const bf16* __restrict__ Q,   // [B*T, C]   (RoPE'd)
    const bf16* __restrict__ K,   // [B*T, kvC] (RoPE'd)
    const bf16* __restrict__ V,   // [B*T, kvC]
    float* __restrict__ Ctx) {    // [B*T, C] in (b,t,h,d) order
  __shared__ bf16 Vt[D_ * 32];        // V tile transposed [d][key], 4 KB
  __shared__ bf16 Pb[4][16 * 32];     // per-wave P scratch (D->A relayout), 4 KB

  const int lane = threadIdx.x & 31;
  const int wave = threadIdx.x >> 5;
  const bool lo  = lane < 16;
  const int ln   = lane & 15;

  const int qtPerSeq = T_ / 16;                       // 128
  int bi = blockIdx.x;
  const int qt  = bi % qtPerSeq;  bi /= qtPerSeq;
  const int kvh = bi % HKV_;      bi /= HKV_;
  const int b   = bi;
  const int qb  = qt * 16;
  const int hq  = kvh * G_ + wave;                    // this wave's q-head

  // Q A-fragments for dims 0..31 and 32..63 (loaded once, reused all tiles)
  const size_t rowQ = (size_t)(b * T_ + qb + ln) * C_ + hq * D_;
  V16 qa[2];
  #pragma unroll
  for (int f = 0; f < 2; ++f) {
    qa[f].h[0] = *(const v8bf*)(Q + rowQ + f * 32 + (lo ? 0 : 16));
    qa[f].h[1] = *(const v8bf*)(Q + rowQ + f * 32 + (lo ? 8 : 24));
  }

  const float scale = 0.125f;                          // 1/sqrt(64)
  float mrow[8], lrow[8];
  #pragma unroll
  for (int r = 0; r < 8; ++r) { mrow[r] = -3.0e38f; lrow[r] = 0.0f; }
  v8f acc[4] = {};

  const int nTiles = (qb + 16 + 31) >> 5;              // causal: keys <= qb+15
  for (int kt = 0; kt < nTiles; ++kt) {
    const int kb = kt * 32;

    __syncthreads();                                   // Vt reuse fence
    for (int e = threadIdx.x; e < 32 * D_; e += 128) { // stage V^T in LDS
      int key = e >> 6, d = e & 63;
      Vt[d * 32 + key] = V[(size_t)(b * T_ + kb + key) * KVC_ + kvh * D_ + d];
    }
    __syncthreads();

    // S = Q K^T (16x32) as two 16-col fragments; K B-frag read straight from
    // row-major K (16 contiguous dims per lane).
    v8f s[2];
    #pragma unroll
    for (int sf = 0; sf < 2; ++sf) {
      v8f z = {};
      #pragma unroll
      for (int f = 0; f < 2; ++f) {
        const size_t rowK = (size_t)(b * T_ + kb + sf * 16 + ln) * KVC_
                          + kvh * D_ + f * 32 + (lo ? 0 : 16);
        v16bf kB = *(const v16bf*)(K + rowK);
        z = __builtin_amdgcn_wmma_f32_16x16x32_bf16(
            false, qa[f].v, false, kB, (short)0, z, false, false);
      }
      s[sf] = z;
    }

    // scale, causal mask, row max (reduction stays inside each 16-lane half,
    // which is exactly one row set of the D-layout)
    float rmax[8];
    #pragma unroll
    for (int r = 0; r < 8; ++r) {
      const int qi = qb + r + (lo ? 0 : 8);
      #pragma unroll
      for (int sf = 0; sf < 2; ++sf) {
        const int kj = kb + sf * 16 + ln;
        float v = s[sf][r] * scale;
        if (kj > qi) v = -3.0e38f;
        s[sf][r] = v;
      }
      rmax[r] = fmaxf(s[0][r], s[1][r]);
    }
    #pragma unroll
    for (int off = 8; off >= 1; off >>= 1)
      #pragma unroll
      for (int r = 0; r < 8; ++r)
        rmax[r] = fmaxf(rmax[r], __shfl_xor(rmax[r], off, 32));

    float alpha[8], rsum[8];
    #pragma unroll
    for (int r = 0; r < 8; ++r) {
      float mNew = fmaxf(mrow[r], rmax[r]);
      alpha[r] = __expf(mrow[r] - mNew);
      mrow[r] = mNew;
      float p0 = __expf(s[0][r] - mNew);
      float p1 = __expf(s[1][r] - mNew);
      s[0][r] = p0; s[1][r] = p1;
      rsum[r] = p0 + p1;
    }
    #pragma unroll
    for (int off = 8; off >= 1; off >>= 1)
      #pragma unroll
      for (int r = 0; r < 8; ++r)
        rsum[r] += __shfl_xor(rsum[r], off, 32);
    #pragma unroll
    for (int r = 0; r < 8; ++r)
      lrow[r] = lrow[r] * alpha[r] + rsum[r];

    // P: D-layout -> A-layout via per-wave LDS buffer (LDS in-order per wave)
    bf16* pb = &Pb[wave][0];
    #pragma unroll
    for (int sf = 0; sf < 2; ++sf)
      #pragma unroll
      for (int r = 0; r < 8; ++r)
        pb[(r + (lo ? 0 : 8)) * 32 + sf * 16 + ln] = f2bf(s[sf][r]);
    asm volatile("s_wait_dscnt 0" ::: "memory");

    V16 pa;
    pa.h[0] = *(const v8bf*)(pb + ln * 32 + (lo ? 0 : 16));
    pa.h[1] = *(const v8bf*)(pb + ln * 32 + (lo ? 8 : 24));

    // acc = acc*alpha + P·V ; V B-frag from transposed LDS tile
    #pragma unroll
    for (int nt = 0; nt < 4; ++nt) {
      v8f cIn;
      #pragma unroll
      for (int r = 0; r < 8; ++r) cIn[r] = acc[nt][r] * alpha[r];
      v16bf vB = *(const v16bf*)(&Vt[(nt * 16 + ln) * 32 + (lo ? 0 : 16)]);
      acc[nt] = __builtin_amdgcn_wmma_f32_16x16x32_bf16(
          false, pa.v, false, vB, (short)0, cIn, false, false);
    }
  }

  // epilogue: normalize and store into (b, t, h, d)
  #pragma unroll
  for (int nt = 0; nt < 4; ++nt)
    #pragma unroll
    for (int r = 0; r < 8; ++r) {
      int m = qb + r + (lo ? 0 : 8);
      Ctx[((size_t)(b * T_ + m) * HQ_ + hq) * D_ + nt * 16 + ln] =
          acc[nt][r] / lrow[r];
    }
}

// ------------------------------ launcher -----------------------------------
extern "C" void kernel_launch(void* const* d_in, const int* in_sizes, int n_in,
                              void* d_out, int out_size, void* d_ws, size_t ws_size,
                              hipStream_t stream) {
  const float* x  = (const float*)d_in[0];
  const float* Wq = (const float*)d_in[1];
  const float* Wk = (const float*)d_in[2];
  const float* Wv = (const float*)d_in[3];
  const float* Wo = (const float*)d_in[4];
  float* out = (float*)d_out;

  const size_t BT = (size_t)B_ * T_;
  char* p = (char*)d_ws;
  bf16*  xb   = (bf16*)p;  p += BT * C_ * 2;
  bf16*  Wqb  = (bf16*)p;  p += (size_t)C_ * C_ * 2;
  bf16*  Wkb  = (bf16*)p;  p += (size_t)KVC_ * C_ * 2;
  bf16*  Wvb  = (bf16*)p;  p += (size_t)KVC_ * C_ * 2;
  bf16*  Wob  = (bf16*)p;  p += (size_t)C_ * C_ * 2;
  float* Qf   = (float*)p; p += BT * C_ * 4;
  float* Kf   = (float*)p; p += BT * KVC_ * 4;
  float* Vf   = (float*)p; p += BT * KVC_ * 4;
  bf16*  Qb   = (bf16*)p;  p += BT * C_ * 2;
  bf16*  Kb   = (bf16*)p;  p += BT * KVC_ * 2;
  bf16*  Vb   = (bf16*)p;  p += BT * KVC_ * 2;
  float* Ctxf = (float*)p; p += BT * C_ * 4;
  bf16*  Ctxb = (bf16*)p;  p += BT * C_ * 2;

  auto cvt = [&](const float* src, bf16* dst, size_t n) {
    cvt_f32_bf16<<<(int)((n + 255) / 256), 256, 0, stream>>>(src, dst, (int)n);
  };

  // 1) precision downconvert for the matrix units
  cvt(x,  xb,  BT * C_);
  cvt(Wq, Wqb, (size_t)C_ * C_);
  cvt(Wk, Wkb, (size_t)KVC_ * C_);
  cvt(Wv, Wvb, (size_t)KVC_ * C_);
  cvt(Wo, Wob, (size_t)C_ * C_);

  // 2) QKV projections (WMMA GEMMs with TDM-staged weight tiles)
  dim3 blk(256);
  gemm_xWt_bf16<<<dim3(BT / 128, C_ / 64),   blk, 0, stream>>>(xb, Wqb, Qf, (int)BT, C_,   C_);
  gemm_xWt_bf16<<<dim3(BT / 128, KVC_ / 64), blk, 0, stream>>>(xb, Wkb, Kf, (int)BT, KVC_, C_);
  gemm_xWt_bf16<<<dim3(BT / 128, KVC_ / 64), blk, 0, stream>>>(xb, Wvb, Vf, (int)BT, KVC_, C_);

  // 3) RoPE on Q,K (fused with bf16 convert); plain convert for V
  {
    size_t nq = BT * (C_ / 2);
    rope_cvt<<<(int)((nq + 255) / 256), 256, 0, stream>>>(Qf, Qb, (int)BT, C_);
    size_t nk = BT * (KVC_ / 2);
    rope_cvt<<<(int)((nk + 255) / 256), 256, 0, stream>>>(Kf, Kb, (int)BT, KVC_);
    cvt(Vf, Vb, BT * KVC_);
  }

  // 4) causal flash GQA: one block per (b, kv-head, 16 queries)
  flash_gqa<<<B_ * HKV_ * (T_ / 16), 128, 0, stream>>>(Qb, Kb, Vb, Ctxf);

  // 5) output projection
  cvt(Ctxf, Ctxb, BT * C_);
  gemm_xWt_bf16<<<dim3(BT / 128, C_ / 64), blk, 0, stream>>>(Ctxb, Wob, out, (int)BT, C_, C_);
}
